// PEGLayer_34308198761093
// MI455X (gfx1250) — matrix operand order repaired and checked by reading, not compiled, not run on hardware
//
#include <hip/hip_runtime.h>
#include <math.h>

#define N_NODES 20000
#define N_EDGES 320000
#define D 256
#define ET 64   // edges per workgroup in the fused MLP

typedef __attribute__((ext_vector_type(16))) __bf16 v16bf;
typedef __attribute__((ext_vector_type(8)))  float  v8f;

union Frag { v16bf v; unsigned u[8]; };

__device__ __forceinline__ float lrelu(float h) { return h >= 0.f ? h : 0.2f * h; }

// 16-bit A-matrix 16x32 lane layout (ISA 7.12.2): pair i (elements e=2i,2i+1):
//   k = e + 8*half        (e < 8)
//   k = e + 8 + 8*half    (e >= 8)
__device__ __forceinline__ int a_k(int i, int half) {
  int e = 2 * i;
  return (e < 8 ? e : e + 8) + 8 * half;
}
// 16-bit B-matrix 32x16: same k order as A (k = e + 8*half for e < 8, e + 8 + 8*half for e >= 8).
__device__ __forceinline__ int b_k(int i, int half) { int e = 2 * i; return (e < 8 ? e : e + 8) + 8 * half; }

// ---------------------------------------------------------------------------
// prep: fold BN into per-feature scale/bias; transpose+convert w2,w3 to bf16
// ---------------------------------------------------------------------------
__global__ void prep_kernel(const float* __restrict__ w1, const float* __restrict__ b1,
                            const float* __restrict__ g1, const float* __restrict__ be1,
                            const float* __restrict__ w2, const float* __restrict__ b2,
                            const float* __restrict__ g2, const float* __restrict__ be2,
                            const float* __restrict__ w3, const float* __restrict__ b3,
                            const float* __restrict__ g3, const float* __restrict__ be3,
                            float* __restrict__ s1, float* __restrict__ t1,
                            float* __restrict__ s2, float* __restrict__ t2,
                            float* __restrict__ s3, float* __restrict__ t3,
                            __bf16* __restrict__ w2t, __bf16* __restrict__ w3t)
{
  const float inv = rsqrtf(1.0f + 1e-5f);
  int i = blockIdx.x * 256 + threadIdx.x;
  if (i < 256) {
    s1[i] = g1[i] * inv * w1[i];
    t1[i] = g1[i] * inv * b1[i] + be1[i];
    s3[i] = g3[i] * inv;
    t3[i] = s3[i] * b3[i] + be3[i];
  }
  if (i < 512) {
    s2[i] = g2[i] * inv;
    t2[i] = s2[i] * b2[i] + be2[i];
  }
  if (i < 256 * 512) {
    { // w2t[n][k] = w2[k][n],  n<512, k<256, row length 256
      int n = i >> 8, k = i & 255;
      w2t[i] = (__bf16)w2[k * 512 + n];
    }
    { // w3t[n][k] = w3[k][n],  n<256, k<512, row length 512
      int n = i >> 9, k = i & 511;
      w3t[i] = (__bf16)w3[k * 256 + n];
    }
  }
}

// ---------------------------------------------------------------------------
// degree / dinv
// ---------------------------------------------------------------------------
__global__ void deg_kernel(const int* __restrict__ ei, float* __restrict__ deg)
{
  int i = blockIdx.x * 256 + threadIdx.x;
  if (i < N_EDGES) atomicAdd(&deg[ei[i]], 1.0f);
}

__global__ void dinv_kernel(const float* __restrict__ deg, float* __restrict__ dinv)
{
  int i = blockIdx.x * 256 + threadIdx.x;
  if (i < N_NODES) {
    float d = deg[i];
    if (d == 0.f) d = 1.f;
    dinv[i] = rsqrtf(d);
  }
}

// ---------------------------------------------------------------------------
// x_proj = x @ Wn + b    (WMMA bf16, 32M x 64N tile per 256-thread WG)
// ---------------------------------------------------------------------------
__global__ __launch_bounds__(256) void xproj_kernel(const float* __restrict__ x,
                                                    const float* __restrict__ Wn,
                                                    const float* __restrict__ bias,
                                                    float* __restrict__ xproj)
{
  __shared__ __align__(16) __bf16 As[32][34];   // [m][k]
  __shared__ __align__(16) __bf16 Bt[64][34];   // [n][k] (transposed)
  const int tid  = threadIdx.x;
  const int wave = tid >> 5, lane = tid & 31;
  const int half = lane >> 4;
  const int mBase = blockIdx.x * 32;
  const int nBase = blockIdx.y * 64;
  const int mt = (wave & 1) * 16;
  const int nt = (wave >> 1) * 16;

  v8f acc = {};
  for (int kk = 0; kk < D; kk += 32) {
    __syncthreads();
    { // stage A: rows mBase..+32, cols kk..+32 -> bf16
      int r  = tid >> 3;
      int c4 = (tid & 7) * 4;
      int gm = mBase + r;
      float4 v = (gm < N_NODES) ? *(const float4*)(x + (size_t)gm * D + kk + c4)
                                : make_float4(0.f, 0.f, 0.f, 0.f);
      As[r][c4 + 0] = (__bf16)v.x; As[r][c4 + 1] = (__bf16)v.y;
      As[r][c4 + 2] = (__bf16)v.z; As[r][c4 + 3] = (__bf16)v.w;
    }
    { // stage B transposed: Wn[kk+k][nBase+n] -> Bt[n][k]
      int k  = tid & 31;
      int ng = (tid >> 5) * 8;
      const float* src = Wn + (size_t)(kk + k) * D + nBase + ng;
      float4 a = *(const float4*)(src);
      float4 b = *(const float4*)(src + 4);
      Bt[ng + 0][k] = (__bf16)a.x; Bt[ng + 1][k] = (__bf16)a.y;
      Bt[ng + 2][k] = (__bf16)a.z; Bt[ng + 3][k] = (__bf16)a.w;
      Bt[ng + 4][k] = (__bf16)b.x; Bt[ng + 5][k] = (__bf16)b.y;
      Bt[ng + 6][k] = (__bf16)b.z; Bt[ng + 7][k] = (__bf16)b.w;
    }
    __syncthreads();
    Frag a, b;
    int m = mt + (lane & 15);
    int n = nt + (lane & 15);
#pragma unroll
    for (int i = 0; i < 8; ++i) a.u[i] = *(const unsigned*)&As[m][a_k(i, half)];
#pragma unroll
    for (int i = 0; i < 8; ++i) b.u[i] = *(const unsigned*)&Bt[n][b_k(i, half)];
    acc = __builtin_amdgcn_wmma_f32_16x16x32_bf16(false, a.v, false, b.v,
                                                  (short)0, acc, false, false);
  }
  int gn = nBase + nt + (lane & 15);
  float bv = bias[gn];
#pragma unroll
  for (int r = 0; r < 8; ++r) {
    int gm = mBase + mt + r + 8 * half;
    if (gm < N_NODES) xproj[(size_t)gm * D + gn] = acc[r] + bv;
  }
}

// ---------------------------------------------------------------------------
// fused edge MLP: 64 edges per WG, hidden layer processed in 4 chunks of 128
// so the [64x512] intermediate never leaves LDS:
//   H1[64x256] (affine+leaky of dist)
//   for chunk c: H2c[64x128] = leaky(bn(H1@w2[:,c])) ; H3acc += H2c@w3[c,:]
//   H3 = leaky(bn(H3acc)) (into H1 buffer) ; emb = H3@w4+b4
//   wedge = dinv[row]*dinv[col]*emb
// 64-edge tile halves L2 weight traffic vs a 32-edge tile (w2t/w3t re-read
// once per WG: 5000 WGs x 512KB = 2.6GB of L2-resident reads).
// ---------------------------------------------------------------------------
__global__ __launch_bounds__(256) void edge_mlp_kernel(
    const int* __restrict__ ei, const float* __restrict__ pos,
    const float* __restrict__ dinv,
    const float* __restrict__ s1, const float* __restrict__ t1,
    const __bf16* __restrict__ w2t, const float* __restrict__ s2, const float* __restrict__ t2,
    const __bf16* __restrict__ w3t, const float* __restrict__ s3, const float* __restrict__ t3,
    const float* __restrict__ w4, const float* __restrict__ b4,
    float* __restrict__ wedge)
{
  __shared__ __align__(16) __bf16 H1[ET][264];    // layer-2 input; later holds H3
  __shared__ __align__(16) __bf16 H2c[ET][136];   // one 128-wide hidden chunk
  __shared__ float distS[ET], normS[ET];
  __shared__ float red[256];

  const int tid  = threadIdx.x;
  const int wave = tid >> 5, lane = tid & 31;
  const int half = lane >> 4;
  const int lm   = lane & 15;
  const int eBase = blockIdx.x * ET;

  if (tid < ET) {
    int e = eBase + tid;
    int r = ei[e], c = ei[N_EDGES + e];
    float dx = pos[r * 3 + 0] - pos[c * 3 + 0];
    float dy = pos[r * 3 + 1] - pos[c * 3 + 1];
    float dz = pos[r * 3 + 2] - pos[c * 3 + 2];
    distS[tid] = sqrtf(dx * dx + dy * dy + dz * dz);
    normS[tid] = dinv[r] * dinv[c];
  }
  __syncthreads();

  { // H1: layer1 folded into scale/bias of dist
    int e  = tid >> 2;
    int j0 = (tid & 3) * 64;
    float d = distS[e];
#pragma unroll
    for (int j = 0; j < 64; ++j) {
      int jj = j0 + j;
      H1[e][jj] = (__bf16)lrelu(s1[jj] * d + t1[jj]);
    }
  }

  // persistent H3 accumulators: wave owns jobs wave*8+j  (j=0..7)
  v8f zero = {};
  v8f acc3[8];
#pragma unroll
  for (int j = 0; j < 8; ++j) acc3[j] = zero;

  for (int c = 0; c < 4; ++c) {
    __syncthreads();  // H1 ready (first iter) / H2c reads from prev chunk done

    // ---- H2 chunk: M=64, N=128, K=256 -> 32 jobs, 4 per wave
#pragma unroll
    for (int j = 0; j < 4; ++j) {
      int job = wave * 4 + j;
      int mt  = (job & 3) * 16;
      int nL  = (job >> 2) * 16;        // 0..112 within chunk
      int gn  = c * 128 + nL + lm;      // global hidden index 0..511
      int m   = mt + lm;
      v8f acc = zero;
      for (int kk = 0; kk < 256; kk += 32) {
        Frag a, b;
#pragma unroll
        for (int i = 0; i < 8; ++i)
          a.u[i] = *(const unsigned*)&H1[m][kk + a_k(i, half)];
#pragma unroll
        for (int i = 0; i < 8; ++i)
          b.u[i] = *(const unsigned*)(w2t + (size_t)gn * 256 + kk + b_k(i, half));
        acc = __builtin_amdgcn_wmma_f32_16x16x32_bf16(false, a.v, false, b.v,
                                                      (short)0, acc, false, false);
      }
      float sv = s2[gn], tv = t2[gn];
#pragma unroll
      for (int r = 0; r < 8; ++r)
        H2c[mt + r + 8 * half][nL + lm] = (__bf16)lrelu(sv * acc[r] + tv);
    }
    __syncthreads();

    // ---- H3 accumulate: M=64, N=256, K=128 (this chunk) -> 64 jobs, 8/wave
#pragma unroll
    for (int j = 0; j < 8; ++j) {
      int job = wave * 8 + j;
      int mt  = (job & 3) * 16;
      int nB  = (job >> 2) * 16;        // 0..240
      int m   = mt + lm;
      int n   = nB + lm;
      if (c < 3)  // hide next chunk's weight latency behind this chunk's WMMAs
        __builtin_prefetch(w3t + (size_t)n * 512 + (c + 1) * 128, 0, 0);
#pragma unroll
      for (int kkl = 0; kkl < 128; kkl += 32) {
        Frag a, b;
#pragma unroll
        for (int i = 0; i < 8; ++i)
          a.u[i] = *(const unsigned*)&H2c[m][kkl + a_k(i, half)];
#pragma unroll
        for (int i = 0; i < 8; ++i)
          b.u[i] = *(const unsigned*)(w3t + (size_t)n * 512 + c * 128 + kkl + b_k(i, half));
        acc3[j] = __builtin_amdgcn_wmma_f32_16x16x32_bf16(false, a.v, false, b.v,
                                                          (short)0, acc3[j], false, false);
      }
    }
  }
  __syncthreads();

  // ---- H3 epilogue: leaky(s3*acc+t3) -> H1 buffer (H1 no longer read)
#pragma unroll
  for (int j = 0; j < 8; ++j) {
    int job = wave * 8 + j;
    int mt  = (job & 3) * 16;
    int nB  = (job >> 2) * 16;
    int gn  = nB + lm;
    float sv = s3[gn], tv = t3[gn];
#pragma unroll
    for (int r = 0; r < 8; ++r)
      H1[mt + r + 8 * half][gn] = (__bf16)lrelu(sv * acc3[j][r] + tv);
  }
  __syncthreads();

  // ---- emb = H3 @ w4 + b4 ; wedge = norm * emb
  {
    int e  = tid >> 2;
    int j0 = (tid & 3) * 64;
    float s = 0.f;
#pragma unroll
    for (int j = 0; j < 64; ++j) s += (float)H1[e][j0 + j] * w4[j0 + j];
    red[tid] = s;
  }
  __syncthreads();
  if (tid < ET) {
    float s = b4[0];
#pragma unroll
    for (int i = 0; i < 4; ++i) s += red[tid * 4 + i];
    wedge[eBase + tid] = normS[tid] * s;
  }
}

// ---------------------------------------------------------------------------
// symmetric scatter: aggr[row]+=w*xproj[col], aggr[col]+=w*xproj[row]
// one WG per edge, one lane per channel; x_proj/aggr are L2-resident (41 MB)
// ---------------------------------------------------------------------------
__global__ __launch_bounds__(256) void scatter_kernel(const int* __restrict__ ei,
                                                      const float* __restrict__ wedge,
                                                      const float* __restrict__ xproj,
                                                      float* __restrict__ aggr)
{
  int e = blockIdx.x;
  int c = threadIdx.x;
  int r  = ei[e];
  int cl = ei[N_EDGES + e];
  float w = wedge[e];
  float xc = xproj[(size_t)cl * D + c];
  float xr = xproj[(size_t)r  * D + c];
  atomicAdd(&aggr[(size_t)r  * D + c], w * xc);
  atomicAdd(&aggr[(size_t)cl * D + c], w * xr);
}

// ---------------------------------------------------------------------------
// out[n] = [relu(aggr[n]), pos[n]]   -> [N, 259]
// ---------------------------------------------------------------------------
__global__ void finalize_kernel(const float* __restrict__ aggr,
                                const float* __restrict__ pos,
                                float* __restrict__ out)
{
  int i = blockIdx.x * 256 + threadIdx.x;
  if (i < N_NODES * 259) {
    int n = i / 259;
    int c = i - n * 259;
    out[i] = (c < D) ? fmaxf(aggr[(size_t)n * D + c], 0.f)
                     : pos[n * 3 + (c - D)];
  }
}

// ---------------------------------------------------------------------------
extern "C" void kernel_launch(void* const* d_in, const int* in_sizes, int n_in,
                              void* d_out, int out_size, void* d_ws, size_t ws_size,
                              hipStream_t stream)
{
  const float* x    = (const float*)d_in[0];
  const float* pos  = (const float*)d_in[1];
  const int*   ei   = (const int*)  d_in[2];
  const float* Wn_w = (const float*)d_in[3];
  const float* Wn_b = (const float*)d_in[4];
  const float* w1   = (const float*)d_in[5];
  const float* b1   = (const float*)d_in[6];
  const float* g1   = (const float*)d_in[7];
  const float* be1  = (const float*)d_in[8];
  const float* w2   = (const float*)d_in[9];
  const float* b2   = (const float*)d_in[10];
  const float* g2   = (const float*)d_in[11];
  const float* be2  = (const float*)d_in[12];
  const float* w3   = (const float*)d_in[13];
  const float* b3   = (const float*)d_in[14];
  const float* g3   = (const float*)d_in[15];
  const float* be3  = (const float*)d_in[16];
  const float* w4   = (const float*)d_in[17];
  const float* b4   = (const float*)d_in[18];

  char* ws = (char*)d_ws;
  size_t o = 0;
  auto alloc = [&](size_t bytes) -> void* {
    void* p = ws + o;
    o = (o + bytes + 255) & ~(size_t)255;
    return p;
  };
  float*  xproj = (float*) alloc((size_t)N_NODES * D * 4);
  float*  aggr  = (float*) alloc((size_t)N_NODES * D * 4);
  float*  deg   = (float*) alloc((size_t)N_NODES * 4);
  float*  dinv  = (float*) alloc((size_t)N_NODES * 4);
  float*  wedge = (float*) alloc((size_t)N_EDGES * 4);
  float*  s1    = (float*) alloc(256 * 4);
  float*  t1    = (float*) alloc(256 * 4);
  float*  s2    = (float*) alloc(512 * 4);
  float*  t2    = (float*) alloc(512 * 4);
  float*  s3    = (float*) alloc(256 * 4);
  float*  t3    = (float*) alloc(256 * 4);
  __bf16* w2t   = (__bf16*)alloc((size_t)256 * 512 * 2);
  __bf16* w3t   = (__bf16*)alloc((size_t)512 * 256 * 2);

  hipMemsetAsync(aggr, 0, (size_t)N_NODES * D * 4, stream);
  hipMemsetAsync(deg,  0, (size_t)N_NODES * 4, stream);

  prep_kernel<<<512, 256, 0, stream>>>(w1, b1, g1, be1, w2, b2, g2, be2,
                                       w3, b3, g3, be3,
                                       s1, t1, s2, t2, s3, t3, w2t, w3t);
  deg_kernel<<<(N_EDGES + 255) / 256, 256, 0, stream>>>(ei, deg);
  dinv_kernel<<<(N_NODES + 255) / 256, 256, 0, stream>>>(deg, dinv);
  xproj_kernel<<<dim3(N_NODES / 32, D / 64), 256, 0, stream>>>(x, Wn_w, Wn_b, xproj);
  edge_mlp_kernel<<<N_EDGES / ET, 256, 0, stream>>>(ei, pos, dinv,
                                                    s1, t1, w2t, s2, t2,
                                                    w3t, s3, t3, w4, b4, wedge);
  scatter_kernel<<<N_EDGES, 256, 0, stream>>>(ei, wedge, xproj, aggr);
  finalize_kernel<<<(N_NODES * 259 + 255) / 256, 256, 0, stream>>>(aggr, pos,
                                                                   (float*)d_out);
}
